// CIGNN_47450798686429
// MI455X (gfx1250) — matrix operand order repaired and checked by reading, not compiled, not run on hardware
//
#include <hip/hip_runtime.h>
#include <cstdint>

#define NB_B     8192
#define NB_K     32
#define NB_D     128

typedef __bf16        v16bf __attribute__((ext_vector_type(16)));
typedef float         v8f   __attribute__((ext_vector_type(8)));
typedef unsigned int  v4u   __attribute__((ext_vector_type(4)));

// ---------------- WMMA helpers -------------------------------------------

static __device__ inline v8f wmma_bf16(v16bf a, v16bf b, v8f c) {
    return __builtin_amdgcn_wmma_f32_16x16x32_bf16(
        false, a, false, b, (short)0, c, false, false);
}

// A-fragment (16x32 bf16) from row-major bf16 matrix.
// Lane L holds row m0+(L&15); K chunks: lanes<16 -> {0-7,16-23}, lanes>=16 -> {8-15,24-31}.
static __device__ inline v16bf ldA(const __bf16* base, int ld, int m0, int k0, int lane) {
    const __bf16* p = base + (size_t)(m0 + (lane & 15)) * ld + k0 + ((lane & 16) ? 8 : 0);
    v16bf a;
    *((v4u*)&a)     = *((const v4u*)p);        // K slots 0..7  (8 bf16 = 16B)
    *(((v4u*)&a)+1) = *((const v4u*)(p + 16)); // K slots 8..15
    return a;
}

// A-fragment gathered from row-major f32 matrix via an index array, cvt->bf16.
static __device__ inline v16bf ldA_gather(const float* mat, const int* idx,
                                          int m0, int k0, int lane) {
    const int row = idx[m0 + (lane & 15)];
    const float* p = mat + (size_t)row * NB_D + k0 + ((lane & 16) ? 8 : 0);
    v16bf a;
#pragma unroll
    for (int i = 0; i < 8; ++i) { a[i] = (__bf16)p[i]; a[8 + i] = (__bf16)p[16 + i]; }
    return a;
}

// B-fragment from pre-packed weights: tile = 32 lanes x 16 bf16, lane-contiguous.
static __device__ inline v16bf ldB(const __bf16* packed, int tile, int lane) {
    const __bf16* p = packed + (size_t)tile * 512 + lane * 16;
    v16bf b;
    *((v4u*)&b)     = *((const v4u*)p);
    *(((v4u*)&b)+1) = *((const v4u*)(p + 8));
    return b;
}

// ---------------- prep kernels -------------------------------------------

// v1 = W_q @ a[:128], v2 = W_k @ a[128:]
__global__ void prep_va(const float* __restrict__ Wq, const float* __restrict__ Wk,
                        const float* __restrict__ a,
                        float* __restrict__ v1, float* __restrict__ v2) {
    int d = threadIdx.x;
    float s1 = 0.f, s2 = 0.f;
    for (int e = 0; e < NB_D; ++e) {
        s1 += Wq[d * NB_D + e] * a[e];
        s2 += Wk[d * NB_D + e] * a[NB_D + e];
    }
    v1[d] = s1; v2[d] = s2;
}

// Pack f32 [Kdim, Nsrc] row-major weight into bf16 B-fragment tiles (Kt x Nt
// tiles of 32x16). Lane L slot v holds W[kt*32 + (L<16?0:8) + (v<8 ? v : 8+v), nt*16 + (L&15)].
__global__ void pack_w(const float* __restrict__ W, __bf16* __restrict__ out,
                       int Kt, int Nt, int Nsrc) {
    int tid = blockIdx.x * blockDim.x + threadIdx.x;
    if (tid >= Kt * Nt * 512) return;
    int t = tid >> 9, within = tid & 511;
    int L = within >> 4, v = within & 15;
    int kt = t / Nt, nt = t % Nt;
    int k = kt * 32 + ((L < 16) ? 0 : 8) + ((v < 8) ? v : 8 + v);
    int n = nt * 16 + (L & 15);
    float val = (n < Nsrc) ? W[(size_t)k * Nsrc + n] : 0.f;
    out[tid] = (__bf16)val;
}

// ---------------- attention (async gather + softmax pool) -----------------
// One block per batch element. ~270MB of gathers dominate the whole workload,
// so neighbor rows go global->LDS on the CDNA5 async path (no VGPR staging),
// tracked by ASYNCcnt.
__global__ void attend_k(const float* __restrict__ mem, const int* __restrict__ idxs,
                         const int* __restrict__ nbrs,
                         const float* __restrict__ v1, const float* __restrict__ v2,
                         __bf16* __restrict__ F, __bf16* __restrict__ P) {
    __shared__ __align__(16) float nb[NB_K * 132];   // padded rows (bank spread)
    __shared__ float selfr[NB_D];
    __shared__ float attn[NB_K];
    const int b = blockIdx.x;
    const int t = threadIdx.x;   // 128 threads

    {   // gather 32 neighbor rows: async global->LDS, 16B per lane per row
        int sub = t >> 5, l4 = t & 31;
#pragma unroll
        for (int g = 0; g < 8; ++g) {
            int k = g * 4 + sub;
            size_t r = (size_t)nbrs[b * NB_K + k];
            const float* src = mem + r * NB_D + l4 * 4;
            // low 32 bits of a flat LDS pointer == LDS byte offset (aperture rule)
            unsigned lds_off = (unsigned)(uintptr_t)(&nb[k * 132 + l4 * 4]);
            asm volatile("global_load_async_to_lds_b128 %0, %1, off"
                         :: "v"(lds_off), "v"(src) : "memory");
        }
    }
    {   // self embedding: to LDS + bf16 into feature cols [0,128)
        size_t r = (size_t)idxs[b];
        float v = mem[r * NB_D + t];
        selfr[t] = v;
        F[(size_t)b * 512 + t] = (__bf16)v;
    }
    asm volatile("s_wait_asynccnt 0x0" ::: "memory");
    __syncthreads();
    if (t < NB_K) {   // wave 0: scores + softmax over K=32
        float sq = 0.f, sk = 0.f;
        for (int c = 0; c < NB_D; ++c) {
            sq += selfr[c] * v1[c];
            sk += nb[t * 132 + c] * v2[c];
        }
        float s = sq + sk;
        s = (s > 0.f) ? s : 0.2f * s;           // leaky_relu
        float m = s;
        for (int off = 16; off; off >>= 1) m = fmaxf(m, __shfl_xor(m, off, 32));
        float e = __expf(s - m);
        float sum = e;
        for (int off = 16; off; off >>= 1) sum += __shfl_xor(sum, off, 32);
        attn[t] = e / sum;
    }
    __syncthreads();
    {   // pooled = attn @ nb  (agg = pooled @ W_k done by WMMA later)
        float p = 0.f;
#pragma unroll
        for (int k = 0; k < NB_K; ++k) p += attn[k] * nb[k * 132 + t];
        P[(size_t)b * NB_D + t] = (__bf16)p;
    }
}

// ---------------- agg = pooled @ W_k  (WMMA) ------------------------------
__global__ void agg_gemm(const __bf16* __restrict__ P, const __bf16* __restrict__ WkP,
                         __bf16* __restrict__ Fa, __bf16* __restrict__ Fb) {
    int lane = threadIdx.x & 31, wave = threadIdx.x >> 5;
    int tile = blockIdx.x * 4 + wave;         // 4096 tiles
    int m0 = (tile >> 3) * 16, nt = tile & 7;
    v8f acc = {};
#pragma unroll
    for (int kt = 0; kt < 4; ++kt)
        acc = wmma_bf16(ldA(P, NB_D, m0, kt * 32, lane), ldB(WkP, kt * 8 + nt, lane), acc);
    int col = nt * 16 + (lane & 15);
    int rbase = m0 + ((lane & 16) ? 8 : 0);
#pragma unroll
    for (int r = 0; r < 8; ++r) {
        __bf16 v = (__bf16)acc[r];
        size_t row = (size_t)(rbase + r);
        Fa[row * 512 + 128 + col] = v;   // own-side agg
        Fb[row * 512 + 256 + col] = v;   // other-side view
    }
}

// ---------------- e = edge_feat[eidx] @ We_w + We_b  (gathered WMMA) ------
__global__ void edge_gemm(const float* __restrict__ EF, const int* __restrict__ eidx,
                          const __bf16* __restrict__ WewP, const float* __restrict__ bias,
                          __bf16* __restrict__ Fs, __bf16* __restrict__ Fd) {
    int lane = threadIdx.x & 31, wave = threadIdx.x >> 5;
    int tile = blockIdx.x * 4 + wave;
    int m0 = (tile >> 3) * 16, nt = tile & 7;
    v8f acc = {};
#pragma unroll
    for (int kt = 0; kt < 4; ++kt)
        acc = wmma_bf16(ldA_gather(EF, eidx, m0, kt * 32, lane),
                        ldB(WewP, kt * 8 + nt, lane), acc);
    int col = nt * 16 + (lane & 15);
    float bv = bias[col];
    int rbase = m0 + ((lane & 16) ? 8 : 0);
#pragma unroll
    for (int r = 0; r < 8; ++r) {
        __bf16 v = (__bf16)(acc[r] + bv);
        size_t row = (size_t)(rbase + r);
        Fs[row * 512 + 384 + col] = v;
        Fd[row * 512 + 384 + col] = v;
    }
}

// ---------------- fused head: relu(relu(F@Wuc)@Wc1)@Wc2 -------------------
// 4 waves/block, each wave owns a 16-row tile; intermediates staged in LDS.
__global__ void head_k(const __bf16* __restrict__ F, const __bf16* __restrict__ WucP,
                       const __bf16* __restrict__ Wc1P, const __bf16* __restrict__ Wc2P,
                       float* __restrict__ out) {
    __shared__ __align__(16) __bf16 H1[4][16 * 128];
    __shared__ __align__(16) __bf16 H2[4][16 * 128];
    int lane = threadIdx.x & 31, wave = threadIdx.x >> 5;
    int m0 = (blockIdx.x * 4 + wave) * 16;
    __bf16* h1 = H1[wave];
    __bf16* h2 = H2[wave];
    int col16 = lane & 15;
    int rloc = (lane & 16) ? 8 : 0;

    // stage 1: H1 = relu(F[16,512] @ Wuc[512,128])
    for (int nt = 0; nt < 8; ++nt) {
        v8f acc = {};
#pragma unroll
        for (int kt = 0; kt < 16; ++kt)
            acc = wmma_bf16(ldA(F, 512, m0, kt * 32, lane), ldB(WucP, kt * 8 + nt, lane), acc);
#pragma unroll
        for (int r = 0; r < 8; ++r) {
            float v = acc[r] > 0.f ? acc[r] : 0.f;
            h1[(rloc + r) * 128 + nt * 16 + col16] = (__bf16)v;
        }
    }
    __syncthreads();
    // stage 2: H2 = relu(H1 @ Wc1[128,128])
    for (int nt = 0; nt < 8; ++nt) {
        v8f acc = {};
#pragma unroll
        for (int kt = 0; kt < 4; ++kt)
            acc = wmma_bf16(ldA(h1, 128, 0, kt * 32, lane), ldB(Wc1P, kt * 8 + nt, lane), acc);
#pragma unroll
        for (int r = 0; r < 8; ++r) {
            float v = acc[r] > 0.f ? acc[r] : 0.f;
            h2[(rloc + r) * 128 + nt * 16 + col16] = (__bf16)v;
        }
    }
    __syncthreads();
    // stage 3: logits = H2 @ Wc2[128,60 pad 64]
    for (int nt = 0; nt < 4; ++nt) {
        v8f acc = {};
#pragma unroll
        for (int kt = 0; kt < 4; ++kt)
            acc = wmma_bf16(ldA(h2, 128, 0, kt * 32, lane), ldB(Wc2P, kt * 4 + nt, lane), acc);
        int col = nt * 16 + col16;
        if (col < 60) {
#pragma unroll
            for (int r = 0; r < 8; ++r)
                out[(size_t)(m0 + rloc + r) * 60 + col] = acc[r];
        }
    }
}

// ---------------- launch --------------------------------------------------

extern "C" void kernel_launch(void* const* d_in, const int* in_sizes, int n_in,
                              void* d_out, int out_size, void* d_ws, size_t ws_size,
                              hipStream_t stream) {
    const float* mem       = (const float*)d_in[0];
    const float* edge_feat = (const float*)d_in[1];
    const float* W_q       = (const float*)d_in[2];
    const float* W_k       = (const float*)d_in[3];
    const float* a_vec     = (const float*)d_in[4];
    const float* We_w      = (const float*)d_in[5];
    const float* We_b      = (const float*)d_in[6];
    const float* Wuc       = (const float*)d_in[7];
    const float* Wc1       = (const float*)d_in[8];
    const float* Wc2       = (const float*)d_in[9];
    const int* src_idxs    = (const int*)d_in[10];
    const int* dst_idxs    = (const int*)d_in[11];
    const int* edge_idxs   = (const int*)d_in[12];
    const int* src_nb      = (const int*)d_in[13];
    const int* dst_nb      = (const int*)d_in[14];
    float* out             = (float*)d_out;

    char* w = (char*)d_ws;
    size_t off = 0;
    auto alloc = [&](size_t bytes) -> void* {
        void* p = w + off;
        off = (off + bytes + 511) & ~(size_t)511;
        return p;
    };
    float*  v1   = (float*)alloc(512);
    float*  v2   = (float*)alloc(512);
    __bf16* WkP  = (__bf16*)alloc(32768);                 // 128x128
    __bf16* WewP = (__bf16*)alloc(32768);                 // 128x128
    __bf16* WucP = (__bf16*)alloc(131072);                // 512x128
    __bf16* Wc1P = (__bf16*)alloc(32768);                 // 128x128
    __bf16* Wc2P = (__bf16*)alloc(16384);                 // 128x64 (60 pad)
    __bf16* Fsrc = (__bf16*)alloc((size_t)NB_B * 512 * 2);
    __bf16* Fdst = (__bf16*)alloc((size_t)NB_B * 512 * 2);
    __bf16* Psrc = (__bf16*)alloc((size_t)NB_B * NB_D * 2);
    __bf16* Pdst = (__bf16*)alloc((size_t)NB_B * NB_D * 2);

    prep_va<<<1, 128, 0, stream>>>(W_q, W_k, a_vec, v1, v2);
    pack_w<<<64, 256, 0, stream>>>(W_k,  WkP,  4, 8, 128);
    pack_w<<<64, 256, 0, stream>>>(We_w, WewP, 4, 8, 128);
    pack_w<<<256, 256, 0, stream>>>(Wuc, WucP, 16, 8, 128);
    pack_w<<<64, 256, 0, stream>>>(Wc1,  Wc1P, 4, 8, 128);
    pack_w<<<32, 256, 0, stream>>>(Wc2,  Wc2P, 4, 4, 60);

    attend_k<<<NB_B, 128, 0, stream>>>(mem, src_idxs, src_nb, v1, v2, Fsrc, Psrc);
    attend_k<<<NB_B, 128, 0, stream>>>(mem, dst_idxs, dst_nb, v1, v2, Fdst, Pdst);

    agg_gemm<<<1024, 128, 0, stream>>>(Psrc, WkP, Fsrc, Fdst);
    agg_gemm<<<1024, 128, 0, stream>>>(Pdst, WkP, Fdst, Fsrc);
    edge_gemm<<<1024, 128, 0, stream>>>(edge_feat, edge_idxs, WewP, We_b, Fsrc, Fdst);

    head_k<<<128, 128, 0, stream>>>(Fsrc, WucP, Wc1P, Wc2P, out);
    head_k<<<128, 128, 0, stream>>>(Fdst, WucP, Wc1P, Wc2P, out + (size_t)NB_B * 60);
}
